// LFCA_41463614276137
// MI455X (gfx1250) — matrix operand order repaired and checked by reading, not compiled, not run on hardware
//
#include <hip/hip_runtime.h>
#include <hip/hip_bf16.h>

typedef __attribute__((ext_vector_type(16))) __bf16 bf16x16;
typedef __attribute__((ext_vector_type(8)))  __bf16 bf16x8;
typedef __attribute__((ext_vector_type(4)))  __bf16 bf16x4;
typedef __attribute__((ext_vector_type(8)))  float  f32x8;
typedef __attribute__((ext_vector_type(4)))  int    i32x4;

#define TPB 256

// ---- CDNA5 async global->LDS staging (probe-guarded) -----------------------
#if defined(__has_builtin)
#  if __has_builtin(__builtin_amdgcn_global_load_async_to_lds_b128)
#    define ASYNC_LDS 1
#  endif
#  if __has_builtin(__builtin_amdgcn_s_wait_asynccnt)
#    define HAVE_WAIT_ASYNC 1
#  endif
#endif

// copy 8 bf16 (16B, both sides 16B-aligned) global -> LDS
__device__ inline void cp8e(__bf16* l, const __bf16* g) {
#ifdef ASYNC_LDS
  __builtin_amdgcn_global_load_async_to_lds_b128(
      (__attribute__((address_space(1))) i32x4*)g,
      (__attribute__((address_space(3))) i32x4*)l, 0, 0);
#else
  *(bf16x8*)l = *(const bf16x8*)g;
#endif
}

__device__ inline void async_join() {
#ifdef ASYNC_LDS
#  ifdef HAVE_WAIT_ASYNC
  __builtin_amdgcn_s_wait_asynccnt(0);
#  else
  asm volatile("s_wait_asynccnt 0x0" ::: "memory");
#  endif
#endif
}

// A fragment: 8 bf16 at p, 8 bf16 at p+16 (ISA A-layout K 0-7/16-23 per lane half)
__device__ inline bf16x16 ldA(const __bf16* p) {
  bf16x8 lo = *(const bf16x8*)p;
  bf16x8 hi = *(const bf16x8*)(p + 16);
  return __builtin_shufflevector(lo, hi, 0, 1, 2, 3, 4, 5, 6, 7, 8, 9, 10, 11,
                                 12, 13, 14, 15);
}
// B fragment: 8 bf16 at p, 8 bf16 at p+16 (same K layout as A)
__device__ inline bf16x16 ldB(const __bf16* p) {
  bf16x8 lo = *(const bf16x8*)p;
  bf16x8 hi = *(const bf16x8*)(p + 16);
  return __builtin_shufflevector(lo, hi, 0, 1, 2, 3, 4, 5, 6, 7, 8, 9, 10, 11,
                                 12, 13, 14, 15);
}

__device__ inline f32x8 wmma_bf16(bf16x16 a, bf16x16 b, f32x8 c) {
  return __builtin_amdgcn_wmma_f32_16x16x32_bf16(false, a, false, b, (short)0,
                                                 c, false, false);
}

// ---------------------------------------------------------------------------
// Elementwise / transform kernels
// ---------------------------------------------------------------------------
__global__ __launch_bounds__(TPB) void k_zero_bf16(__bf16* __restrict__ p, long n8) {
  long stride = (long)gridDim.x * TPB;
  bf16x8 z = {};
  for (long i = (long)blockIdx.x * TPB + threadIdx.x; i < n8; i += stride)
    *(bf16x8*)(p + i * 8) = z;
}

__global__ __launch_bounds__(TPB) void k_cast_bf16(const float* __restrict__ in,
                                                   __bf16* __restrict__ out, long n) {
  long stride = (long)gridDim.x * TPB;
  for (long i = (long)blockIdx.x * TPB + threadIdx.x; i < n; i += stride)
    out[i] = (__bf16)in[i];
}

// fp32 NCHW -> zero-padded bf16 [B,C,H+2,W+2] interior
__global__ __launch_bounds__(TPB) void k_cast_pad(const float* __restrict__ in,
                                                  __bf16* __restrict__ out,
                                                  int H, int W, long total) {
  long stride = (long)gridDim.x * TPB;
  const int Wp = W + 2, Hp = H + 2;
  for (long i = (long)blockIdx.x * TPB + threadIdx.x; i < total; i += stride) {
    const int w = (int)(i % W);
    const int h = (int)((i / W) % H);
    const long bc = i / ((long)H * W);
    out[(bc * Hp + h + 1) * Wp + (w + 1)] = (__bf16)in[i];
  }
}

// w[O][C][3][3] -> wt[tap][O][C]  (k=C contiguous for B staging)
__global__ __launch_bounds__(TPB) void k_wtrans3x3(const float* __restrict__ w,
                                                   __bf16* __restrict__ wt,
                                                   int Cin, int Cout) {
  long n = (long)Cout * Cin * 9;
  long stride = (long)gridDim.x * TPB;
  for (long i = (long)blockIdx.x * TPB + threadIdx.x; i < n; i += stride) {
    int s = (int)(i % 3); long t = i / 3;
    int r = (int)(t % 3); t /= 3;
    int c = (int)(t % Cin);
    int o = (int)(t / Cin);
    wt[((long)(r * 3 + s) * Cout + o) * Cin + c] = (__bf16)w[i];
  }
}

// ---------------------------------------------------------------------------
// Universal implicit-GEMM conv (3x3 pad=1 or 1x1), padded bf16 NCHW input.
// Double-buffered LDS, async B staging, pipelined A staging.
// Block: 256 thr = 8 waves; tile 32(M=w-run) x 128(N=outch); K step 32.
// ---------------------------------------------------------------------------
__global__ __launch_bounds__(TPB)
void k_conv(const __bf16* __restrict__ xp,  // [B,Cin,H+2,W+2]
            const __bf16* __restrict__ wt,  // [ntaps][Cout][Cin]
            const float* __restrict__ bias,
            float* __restrict__ outF, long sc_m, long sc_n, long bsc,
            __bf16* __restrict__ outB, int padOutB,
            const float* __restrict__ preAdd,  // before relu (outF strides)
            const float* __restrict__ res,     // after relu  (outF strides)
            int Cin, int Cout, int H, int W, int ntaps, int relu) {
  __shared__ __align__(16) __bf16 As[2][32][48];
  __shared__ __align__(16) __bf16 Bs[2][128][48];

  const int Wp = W + 2, Hp = H + 2;
  const int mtPerRow = W >> 5;
  const int mt = blockIdx.x;
  const int w0 = (mt % mtPerRow) * 32;
  const int h  = (mt / mtPerRow) % H;
  const int b  = mt / (mtPerRow * H);
  const int n0 = blockIdx.y * 128;

  const int tid = threadIdx.x, lane = tid & 31, wave = tid >> 5;
  const int waveM = wave & 1, waveN = wave >> 1;

  const long planeP = (long)Hp * Wp;
  const __bf16* xb = xp + (long)b * Cin * planeP;

  // staging roles
  const int aK = tid >> 3;        // A: k row 0..31
  const int aM = (tid & 7) * 4;   // A: m base (4 contiguous)
  const int bN = tid >> 1;        // B: n row 0..127
  const int bK = (tid & 1) * 16;  // B: k base (16 contiguous)

  // fragment geometry
  const int mrow = (lane & 15) + waveM * 16;
  const int kbA = (lane < 16) ? 0 : 8;
  const int kbB = (lane < 16) ? 0 : 8;
  const int ncf = waveN * 32 + (lane & 15);

  const int kChunks = Cin >> 5;
  const int total = ntaps * kChunks;

  auto aSrc = [&](int ch) -> const __bf16* {
    const int tap = ch / kChunks;
    const int c0 = (ch % kChunks) << 5;
    const int dy = (ntaps == 1) ? 0 : (tap / 3 - 1);
    const int dx = (ntaps == 1) ? 0 : (tap % 3 - 1);
    const long rowOff = (long)(h + dy + 1) * Wp + (w0 + dx + 1);
    return xb + (long)(c0 + aK) * planeP + rowOff + aM;
  };
  auto bSrc = [&](int ch) -> const __bf16* {
    const int tap = ch / kChunks;
    const int c0 = (ch % kChunks) << 5;
    return wt + (long)tap * Cout * Cin + (long)(n0 + bN) * Cin + (c0 + bK);
  };

  f32x8 acc0 = {}, acc1 = {};
  bf16x4 aReg;

  // prologue: stage chunk 0 into buffer 0
  __builtin_memcpy(&aReg, aSrc(0), sizeof(aReg));
  {
    const __bf16* s = bSrc(0);
    cp8e(&Bs[0][bN][bK], s);
    cp8e(&Bs[0][bN][bK + 8], s + 8);
  }
  As[0][aM + 0][aK] = aReg[0];
  As[0][aM + 1][aK] = aReg[1];
  As[0][aM + 2][aK] = aReg[2];
  As[0][aM + 3][aK] = aReg[3];
  async_join();
  __syncthreads();

  for (int ch = 0; ch < total; ++ch) {
    const int cur = ch & 1, nxt = cur ^ 1;
    const bool more = (ch + 1 < total);
    if (more) {  // issue next-chunk staging (no waits yet)
      __builtin_memcpy(&aReg, aSrc(ch + 1), sizeof(aReg));
      const __bf16* s = bSrc(ch + 1);
      cp8e(&Bs[nxt][bN][bK], s);
      cp8e(&Bs[nxt][bN][bK + 8], s + 8);
    }
    // compute current chunk (overlaps async DMA + A load latency)
    const bf16x16 af  = ldA(&As[cur][mrow][kbA]);
    const bf16x16 bf0 = ldB(&Bs[cur][ncf][kbB]);
    const bf16x16 bf1 = ldB(&Bs[cur][ncf + 16][kbB]);
    acc0 = wmma_bf16(af, bf0, acc0);
    acc1 = wmma_bf16(af, bf1, acc1);
    if (more) {  // transpose-store A for next chunk
      As[nxt][aM + 0][aK] = aReg[0];
      As[nxt][aM + 1][aK] = aReg[1];
      As[nxt][aM + 2][aK] = aReg[2];
      As[nxt][aM + 3][aK] = aReg[3];
    }
    async_join();
    __syncthreads();
  }

  const int col = lane & 15;
  const int rh = (lane < 16) ? 0 : 8;
  const long mG0 = (long)h * W + w0;
#pragma unroll
  for (int t = 0; t < 2; ++t) {
    const f32x8 a = t ? acc1 : acc0;
    const int n = n0 + waveN * 32 + t * 16 + col;
#pragma unroll
    for (int v = 0; v < 8; ++v) {
      const int m = waveM * 16 + rh + v;
      float val = a[v];
      if (bias) val += bias[n];
      const long ci = (long)b * bsc + (long)n * sc_n + (mG0 + m) * sc_m;
      if (preAdd) val += preAdd[ci];
      if (relu) val = fmaxf(val, 0.0f);
      if (res) val += res[ci];
      if (outF) outF[ci] = val;
      if (outB) {
        const long oi = padOutB
            ? (((long)b * Cout + n) * Hp + h + 1) * (long)Wp + (w0 + m + 1)
            : ci;
        outB[oi] = (__bf16)val;
      }
    }
  }
}

// ---------------------------------------------------------------------------
// Attention GEMM: A[m][k] (k contiguous, row stride sa_m), B[n][k] (k
// contiguous, row stride sb_n), C = alpha * A x B, fp32 strided out.
// Double-buffered LDS, fully async staging. Tile 32(M) x 128(N); M ragged ok.
// ---------------------------------------------------------------------------
__global__ __launch_bounds__(TPB)
void k_gemm(const __bf16* __restrict__ A, long sa_m, long bsa,
            const __bf16* __restrict__ Bm, long sb_n, long bsb,
            float alpha,
            float* __restrict__ outC, long sc_m, long sc_n, long bsc,
            int M, int N, int K) {
  __shared__ __align__(16) __bf16 As[2][32][48];
  __shared__ __align__(16) __bf16 Bs[2][128][48];

  const int b = blockIdx.z;
  const __bf16* Ab = A + (long)b * bsa;
  const __bf16* Bb = Bm + (long)b * bsb;
  const int m0 = blockIdx.x * 32;
  const int n0 = blockIdx.y * 128;

  const int tid = threadIdx.x, lane = tid & 31, wave = tid >> 5;
  const int waveM = wave & 1, waveN = wave >> 1;

  const int bN = tid >> 1;
  const int bK = (tid & 1) * 16;
  const int am = tid >> 2;        // A: m row (tid<128)
  const int akq = (tid & 3) * 8;  // A: k base

  const int mrow = (lane & 15) + waveM * 16;
  const int kbA = (lane < 16) ? 0 : 8;
  const int kbB = (lane < 16) ? 0 : 8;
  const int ncf = waveN * 32 + (lane & 15);

  const int total = K >> 5;

  auto stage = [&](int ch, int buf) {
    const int k0 = ch << 5;
    if (tid < 128) {
      if (m0 + am < M) {
        cp8e(&As[buf][am][akq], Ab + (long)(m0 + am) * sa_m + (k0 + akq));
      } else {
        bf16x8 z = {};
        *(bf16x8*)&As[buf][am][akq] = z;
      }
    }
    const __bf16* s = Bb + (long)(n0 + bN) * sb_n + (k0 + bK);
    cp8e(&Bs[buf][bN][bK], s);
    cp8e(&Bs[buf][bN][bK + 8], s + 8);
  };

  f32x8 acc0 = {}, acc1 = {};

  stage(0, 0);
  async_join();
  __syncthreads();

  for (int ch = 0; ch < total; ++ch) {
    const int cur = ch & 1, nxt = cur ^ 1;
    if (ch + 1 < total) stage(ch + 1, nxt);
    const bf16x16 af  = ldA(&As[cur][mrow][kbA]);
    const bf16x16 bf0 = ldB(&Bs[cur][ncf][kbB]);
    const bf16x16 bf1 = ldB(&Bs[cur][ncf + 16][kbB]);
    acc0 = wmma_bf16(af, bf0, acc0);
    acc1 = wmma_bf16(af, bf1, acc1);
    async_join();
    __syncthreads();
  }

  const int col = lane & 15;
  const int rh = (lane < 16) ? 0 : 8;
#pragma unroll
  for (int t = 0; t < 2; ++t) {
    const f32x8 a = t ? acc1 : acc0;
    const int n = n0 + waveN * 32 + t * 16 + col;
#pragma unroll
    for (int v = 0; v < 8; ++v) {
      const int m = waveM * 16 + rh + v;
      if (m0 + m >= M) continue;
      outC[(long)b * bsc + (long)(m0 + m) * sc_m + (long)n * sc_n] =
          a[v] * alpha;
    }
  }
}

// ---------------------------------------------------------------------------
// GroupNorm(1) stats + apply (apply writes padded bf16 NCHW)
// ---------------------------------------------------------------------------
__global__ __launch_bounds__(TPB) void k_gn_stats(const float* __restrict__ x,
                                                  float* __restrict__ stats,
                                                  long perB) {
  __shared__ float rs[TPB], rs2[TPB];
  const int b = blockIdx.x, tid = threadIdx.x;
  const float* p = x + (long)b * perB;
  float s = 0.f, s2 = 0.f;
  for (long i = tid; i < perB; i += TPB) { float v = p[i]; s += v; s2 += v * v; }
  rs[tid] = s; rs2[tid] = s2;
  __syncthreads();
  for (int off = TPB / 2; off; off >>= 1) {
    if (tid < off) { rs[tid] += rs[tid + off]; rs2[tid] += rs2[tid + off]; }
    __syncthreads();
  }
  if (tid == 0) {
    float mu = rs[0] / (float)perB;
    float var = rs2[0] / (float)perB - mu * mu;
    stats[b * 2 + 0] = mu;
    stats[b * 2 + 1] = rsqrtf(var + 1e-6f);
  }
}

__global__ __launch_bounds__(TPB) void k_gn_apply_pad(
    const float* __restrict__ x, const float* __restrict__ stats,
    const float* __restrict__ g, const float* __restrict__ be,
    __bf16* __restrict__ out, int C, int H, int W, long total) {
  long stride = (long)gridDim.x * TPB;
  const int Wp = W + 2, Hp = H + 2;
  const long HW = (long)H * W;
  for (long i = (long)blockIdx.x * TPB + threadIdx.x; i < total; i += stride) {
    const int w = (int)(i % W);
    const int h = (int)((i / W) % H);
    const long bc = i / HW;
    const int c = (int)(bc % C);
    const long b = bc / C;
    const float mu = stats[b * 2], rstd = stats[b * 2 + 1];
    out[(bc * Hp + h + 1) * Wp + (w + 1)] =
        (__bf16)((x[i] - mu) * rstd * g[c] + be[c]);
  }
}

// ---------------------------------------------------------------------------
// Row softmax over N=4096, fp32 in -> bf16 probabilities out
// ---------------------------------------------------------------------------
__global__ __launch_bounds__(TPB) void k_softmax_bf16(const float* __restrict__ S,
                                                      __bf16* __restrict__ P) {
  __shared__ float row[4096];
  __shared__ float red[TPB];
  const long r = blockIdx.x;
  const int tid = threadIdx.x;
  const float* src = S + r * 4096;
  float mx = -3.402823466e38f;
  for (int i = tid; i < 4096; i += TPB) { float v = src[i]; row[i] = v; mx = fmaxf(mx, v); }
  red[tid] = mx; __syncthreads();
  for (int off = TPB / 2; off; off >>= 1) {
    if (tid < off) red[tid] = fmaxf(red[tid], red[tid + off]);
    __syncthreads();
  }
  mx = red[0]; __syncthreads();
  float s = 0.f;
  for (int i = tid; i < 4096; i += TPB) { float e = __expf(row[i] - mx); row[i] = e; s += e; }
  red[tid] = s; __syncthreads();
  for (int off = TPB / 2; off; off >>= 1) {
    if (tid < off) red[tid] += red[tid + off];
    __syncthreads();
  }
  const float inv = 1.0f / red[0];
  __bf16* dst = P + r * 4096;
  for (int i = tid; i < 4096; i += TPB) dst[i] = (__bf16)(row[i] * inv);
}

// out_cross[:, 0, :] -> d_out[0..2047]
__global__ __launch_bounds__(TPB) void k_tok_copy(const float* __restrict__ oc,
                                                  float* __restrict__ dst) {
  const int i = blockIdx.x * TPB + threadIdx.x;
  if (i < 8 * 256) {
    const int b = i >> 8, f = i & 255;
    dst[i] = oc[(long)b * 1025 * 256 + f];
  }
}

// out_cross[:,1:,:] (B,1024,256) -> padded bf16 NCHW (B,256,34,34)
__global__ __launch_bounds__(TPB) void k_tok_to_nchw_pad(const float* __restrict__ oc,
                                                         __bf16* __restrict__ out) {
  const long total = 8L * 256 * 1024;
  long stride = (long)gridDim.x * TPB;
  for (long i = (long)blockIdx.x * TPB + threadIdx.x; i < total; i += stride) {
    const int w = (int)(i % 32);
    const int h = (int)((i / 32) % 32);
    const long bc = i / 1024;  // b*256+f
    const int f = (int)(bc % 256);
    const long b = bc / 256;
    out[(bc * 34 + h + 1) * 34 + (w + 1)] =
        (__bf16)oc[(b * 1025 + 1 + (long)h * 32 + w) * 256 + f];
  }
}

// align-corners bilinear 32->64 + add feat_diff (fp32 NCHW out)
__global__ __launch_bounds__(TPB) void k_resize_add(const float* __restrict__ h0,
                                                    const float* __restrict__ fd,
                                                    float* __restrict__ out) {
  const long total = 8L * 256 * 64 * 64;
  long stride = (long)gridDim.x * TPB;
  const float sc = 31.0f / 63.0f;
  for (long i = (long)blockIdx.x * TPB + threadIdx.x; i < total; i += stride) {
    const int ow = (int)(i % 64);
    const int oh = (int)((i / 64) % 64);
    const long bc = i / 4096;
    const float fy = oh * sc;
    int y0 = (int)floorf(fy); float wy = fy - y0; int y1 = min(y0 + 1, 31);
    const float fx = ow * sc;
    int x0 = (int)floorf(fx); float wx = fx - x0; int x1 = min(x0 + 1, 31);
    const float* p = h0 + bc * 1024;
    const float top = p[y0 * 32 + x0] * (1.f - wx) + p[y0 * 32 + x1] * wx;
    const float bot = p[y1 * 32 + x0] * (1.f - wx) + p[y1 * 32 + x1] * wx;
    out[i] = top * (1.f - wy) + bot * wy + fd[i];
  }
}

// ---------------------------------------------------------------------------
// Host-side orchestration
// ---------------------------------------------------------------------------
static inline size_t alignUp(size_t v) { return (v + 255) & ~(size_t)255; }
static inline int nb(long n) { long b = (n + TPB - 1) / TPB; return (int)(b > 65535 ? 65535 : b); }

extern "C" void kernel_launch(void* const* d_in, const int* in_sizes, int n_in,
                              void* d_out, int out_size, void* d_ws, size_t ws_size,
                              hipStream_t stream) {
  (void)in_sizes; (void)n_in; (void)out_size; (void)ws_size;
  const float* q     = (const float*)d_in[0];
  const float* x     = (const float*)d_in[1];
  const float* rb_w1 = (const float*)d_in[2];
  const float* rb_b1 = (const float*)d_in[3];
  const float* rb_w2 = (const float*)d_in[4];
  const float* rb_b2 = (const float*)d_in[5];
  const float* rb_ws = (const float*)d_in[6];
  const float* rb_bs = (const float*)d_in[7];
  const float* ln_g  = (const float*)d_in[8];
  const float* ln_b  = (const float*)d_in[9];
  const float* key_w = (const float*)d_in[10];
  const float* key_b = (const float*)d_in[11];
  const float* val_w = (const float*)d_in[12];
  const float* val_b = (const float*)d_in[13];
  const float* dh0_w = (const float*)d_in[14];
  const float* dh0_b = (const float*)d_in[15];
  const float* dh1_w = (const float*)d_in[16];
  const float* dh1_b = (const float*)d_in[17];

  const long Bn = 8, HW = 4096, Ntok = 1025;
  const long P64 = 66L * 66;
  const long P32 = 34L * 34;

  char* ws = (char*)d_ws;
  size_t off = 0;
  auto take = [&](size_t bytes) { char* p = ws + off; off = alignUp(off + bytes); return p; };

  __bf16* x_bp  = (__bf16*)take(Bn * 1280 * P64 * 2);
  __bf16* q_b   = (__bf16*)take(Bn * Ntok * 256 * 2);
  __bf16* w1t   = (__bf16*)take(9L * 256 * 1280 * 2);
  __bf16* w2t   = (__bf16*)take(9L * 256 * 256 * 2);
  __bf16* wsb   = (__bf16*)take(256L * 1280 * 2);
  __bf16* kwb   = (__bf16*)take(256L * 256 * 2);
  __bf16* vwb   = (__bf16*)take(256L * 256 * 2);
  __bf16* d0t   = (__bf16*)take(9L * 256 * 256 * 2);
  __bf16* d1t   = (__bf16*)take(9L * 256 * 256 * 2);
  __bf16* y1bp  = (__bf16*)take(Bn * 256 * P64 * 2);
  float*  hF    = (float*)take(Bn * 256 * HW * 4);
  float*  fd    = (float*)take(Bn * 256 * HW * 4);
  __bf16* flnp  = (__bf16*)take(Bn * 256 * P64 * 2);
  __bf16* keyb  = (__bf16*)take(Bn * HW * 256 * 2);    // [b][m][d]
  __bf16* valT  = (__bf16*)take(Bn * 256 * HW * 2);    // [b][f][m]
  float*  scoF  = (float*)take(Bn * Ntok * HW * 4);
  __bf16* probB = (__bf16*)take(Bn * Ntok * HW * 2);
  float*  ocF   = (float*)take(Bn * Ntok * 256 * 4);
  __bf16* op32p = (__bf16*)take(Bn * 256 * P32 * 2);
  float*  h0F   = (float*)take(Bn * 256 * 1024 * 4);
  float*  op64F = (float*)take(Bn * 256 * HW * 4);
  __bf16* gn2p  = (__bf16*)take(Bn * 256 * P64 * 2);
  float*  stats = (float*)take(Bn * 2 * 4);

  float* out_tok = (float*)d_out;
  float* out_p   = (float*)d_out + 2048;

  // ---- zero padded buffers (borders must be 0 every call) ----
  k_zero_bf16<<<nb(Bn * 1280 * P64 / 8), TPB, 0, stream>>>(x_bp, Bn * 1280 * P64 / 8);
  k_zero_bf16<<<nb(Bn * 256 * P64 / 8), TPB, 0, stream>>>(y1bp, Bn * 256 * P64 / 8);
  k_zero_bf16<<<nb(Bn * 256 * P64 / 8), TPB, 0, stream>>>(flnp, Bn * 256 * P64 / 8);
  k_zero_bf16<<<nb(Bn * 256 * P32 / 8), TPB, 0, stream>>>(op32p, Bn * 256 * P32 / 8);
  k_zero_bf16<<<nb(Bn * 256 * P64 / 8), TPB, 0, stream>>>(gn2p, Bn * 256 * P64 / 8);

  // ---- conversions / weight transforms ----
  k_cast_pad<<<nb(Bn * 1280 * HW), TPB, 0, stream>>>(x, x_bp, 64, 64, Bn * 1280 * HW);
  k_cast_bf16<<<nb(Bn * Ntok * 256), TPB, 0, stream>>>(q, q_b, Bn * Ntok * 256);
  k_wtrans3x3<<<nb(9L * 1280 * 256), TPB, 0, stream>>>(rb_w1, w1t, 1280, 256);
  k_wtrans3x3<<<nb(9L * 256 * 256), TPB, 0, stream>>>(rb_w2, w2t, 256, 256);
  k_cast_bf16<<<nb(1280L * 256), TPB, 0, stream>>>(rb_ws, wsb, 1280L * 256);
  k_cast_bf16<<<nb(256L * 256), TPB, 0, stream>>>(key_w, kwb, 256L * 256);
  k_cast_bf16<<<nb(256L * 256), TPB, 0, stream>>>(val_w, vwb, 256L * 256);
  k_wtrans3x3<<<nb(9L * 256 * 256), TPB, 0, stream>>>(dh0_w, d0t, 256, 256);
  k_wtrans3x3<<<nb(9L * 256 * 256), TPB, 0, stream>>>(dh1_w, d1t, 256, 256);

  const dim3 g64(2 * 64 * 8, 2);
  const dim3 g32(1 * 32 * 8, 2);

  // ---- residual block ----
  k_conv<<<g64, TPB, 0, stream>>>(x_bp, w1t, rb_b1,
                                  nullptr, 1, HW, 256 * HW,
                                  y1bp, 1, nullptr, nullptr,
                                  1280, 256, 64, 64, 9, 1);
  k_conv<<<g64, TPB, 0, stream>>>(y1bp, w2t, rb_b2,
                                  hF, 1, HW, 256 * HW,
                                  nullptr, 0, nullptr, nullptr,
                                  256, 256, 64, 64, 9, 0);
  k_conv<<<g64, TPB, 0, stream>>>(x_bp, wsb, rb_bs,
                                  fd, 1, HW, 256 * HW,
                                  nullptr, 0, hF, nullptr,
                                  1280, 256, 64, 64, 1, 1);

  // ---- GN1 -> fln (padded bf16) ----
  k_gn_stats<<<8, TPB, 0, stream>>>(fd, stats, 256L * HW);
  k_gn_apply_pad<<<nb(Bn * 256 * HW), TPB, 0, stream>>>(
      fd, stats, ln_g, ln_b, flnp, 256, 64, 64, Bn * 256 * HW);

  // ---- key / value 1x1 convs ----
  k_conv<<<g64, TPB, 0, stream>>>(flnp, kwb, key_b,
                                  nullptr, 256, 1, HW * 256,
                                  keyb, 0, nullptr, nullptr,
                                  256, 256, 64, 64, 1, 0);
  k_conv<<<g64, TPB, 0, stream>>>(flnp, vwb, val_b,
                                  nullptr, 1, HW, HW * 256,
                                  valT, 0, nullptr, nullptr,
                                  256, 256, 64, 64, 1, 0);

  // ---- attention ----
  k_gemm<<<dim3(33, 32, 8), TPB, 0, stream>>>(
      q_b, 256, Ntok * 256, keyb, 256, HW * 256, 0.0625f,
      scoF, HW, 1, Ntok * HW, 1025, 4096, 256);
  k_softmax_bf16<<<(int)(Bn * Ntok), TPB, 0, stream>>>(scoF, probB);
  k_gemm<<<dim3(33, 2, 8), TPB, 0, stream>>>(
      probB, HW, Ntok * HW, valT, HW, 256 * HW, 1.0f,
      ocF, 256, 1, Ntok * 256, 1025, 256, 4096);

  k_tok_copy<<<8, TPB, 0, stream>>>(ocF, out_tok);
  k_tok_to_nchw_pad<<<nb(Bn * 256 * 1024), TPB, 0, stream>>>(ocF, op32p);

  // ---- decoder head ----
  k_conv<<<g32, TPB, 0, stream>>>(op32p, d0t, dh0_b,
                                  h0F, 1, 1024, 256 * 1024,
                                  nullptr, 0, nullptr, nullptr,
                                  256, 256, 32, 32, 9, 1);
  k_resize_add<<<nb(Bn * 256 * HW), TPB, 0, stream>>>(h0F, fd, op64F);
  k_gn_stats<<<8, TPB, 0, stream>>>(op64F, stats, 256L * HW);
  k_gn_apply_pad<<<nb(Bn * 256 * HW), TPB, 0, stream>>>(
      op64F, stats, ln_g, ln_b, gn2p, 256, 64, 64, Bn * 256 * HW);
  k_conv<<<g64, TPB, 0, stream>>>(gn2p, d1t, dh1_b,
                                  out_p, 1, HW, 256 * HW,
                                  nullptr, 0, nullptr, op64F,
                                  256, 256, 64, 64, 9, 1);
}
